// Aligner_38946763440524
// MI455X (gfx1250) — compile-verified, matrix operand order checked by reading, not run on hardware
//
#include <hip/hip_runtime.h>
#include <math.h>

typedef __attribute__((ext_vector_type(8)))  float  v8f;
typedef __attribute__((ext_vector_type(16))) __bf16 v16bf;

#define LDS_STRIDE 40   // ushorts per LDS row: 32 data + 8 pad (80 B, multiple of 16)

enum { AM_ROW = 0, AM_TRANS = 1 };
enum { BM_TRANS = 0, BM_CONV = 1 };
enum { BIAS_NONE = 0, BIAS_N = 1, BIAS_M = 2 };

struct GemmParams {
  const float* A;      // A operand (batch 0)
  const float* Bp;     // B operand (batch 0)
  const float* bias;
  float*       C;
  int M, N, K;
  int lda, ldb, ldc;
  int mshift, Msrc;    // AM_ROW: a(m,k)=A[(m+mshift)*lda+k] if 0<=m+mshift<Msrc else 0
  int Nsrc;            // BM_CONV: valid source length for js=j+t-1
  long long strideA, strideB, strideC;
};

// split f32 into hi/lo bf16 bit patterns (truncation; lo carries the residual)
__device__ __forceinline__ void split1(float f, unsigned &hi, unsigned &lo) {
  unsigned b = __float_as_uint(f);
  hi = b >> 16;
  float r = f - __uint_as_float(hi << 16);
  lo = __float_as_uint(r) >> 16;
}

// load one 16x32 bf16 WMMA operand fragment from LDS ([row][k] layout).
// lane L<16: row=L, K={0..7,16..23}; lane>=16: row=L-16, K={8..15,24..31}
__device__ __forceinline__ v16bf ldfrag(const unsigned short* arr, int row, int ksel) {
  union { uint4 u[2]; v16bf v; } f;
  f.u[0] = *(const uint4*)(arr + row * LDS_STRIDE + ksel);
  f.u[1] = *(const uint4*)(arr + row * LDS_STRIDE + 16 + ksel);
  return f.v;
}

__device__ __forceinline__ v8f wmma_bf16(v16bf a, v16bf b, v8f c) {
  return __builtin_amdgcn_wmma_f32_16x16x32_bf16(false, a, false, b, (short)0, c,
                                                 false, false);
}

template<int AMODE, int BMODE, int BIASMODE>
__global__ __launch_bounds__(256)
void gemm_kernel(GemmParams p) {
  __shared__ __align__(16) unsigned short As_hi[128 * LDS_STRIDE];
  __shared__ __align__(16) unsigned short As_lo[128 * LDS_STRIDE];
  __shared__ __align__(16) unsigned short Bs_hi[128 * LDS_STRIDE];
  __shared__ __align__(16) unsigned short Bs_lo[128 * LDS_STRIDE];

  const int tid = threadIdx.x;
  const int m0  = blockIdx.y * 128;
  const int n0  = blockIdx.x * 128;
  const float* A  = p.A  + (size_t)blockIdx.z * p.strideA;
  const float* Bp = p.Bp + (size_t)blockIdx.z * p.strideB;
  float*       C  = p.C  + (size_t)blockIdx.z * p.strideC;

  const int lane  = tid & 31;
  const int wid   = tid >> 5;
  const int wm    = wid >> 1;           // 0..3 -> 32 rows each
  const int wn    = wid & 1;            // 0..1 -> 64 cols each
  const int row16 = lane & 15;
  const int ksel  = (lane >> 4) * 8;

  v8f acc[2][4];
  #pragma unroll
  for (int i = 0; i < 2; ++i)
    #pragma unroll
    for (int j = 0; j < 4; ++j)
      acc[i][j] = (v8f){0.f,0.f,0.f,0.f,0.f,0.f,0.f,0.f};

  const int nk = (p.K + 31) >> 5;

  // register-staged tiles (software pipeline: fetch kt+1 during compute of kt)
  float4 stA[4];
  float4 stB[4];      // BM_TRANS staging
  float  stBc[16];    // BM_CONV staging

  auto fetchA = [&](int k0) {
    if (AMODE == AM_ROW) {
      #pragma unroll
      for (int g = 0; g < 4; ++g) {
        int lin = tid + g * 256;        // 0..1023
        int m   = lin >> 3;
        int kg  = (lin & 7) << 2;       // 0,4,...,28
        int k   = k0 + kg;
        int ms  = m0 + m + p.mshift;
        float4 v = make_float4(0.f, 0.f, 0.f, 0.f);
        if (ms >= 0 && ms < p.Msrc && k < p.K) {
          const float* src = A + (size_t)ms * p.lda + k;
          v = *(const float4*)src;
          if (g == 0 && k + 64 < p.K) __builtin_prefetch(src + 64, 0, 1);
        }
        stA[g] = v;
      }
    } else {  // AM_TRANS: a(m,k) = A[k*lda + m]  (m-contiguous source)
      #pragma unroll
      for (int g = 0; g < 4; ++g) {
        int lin = tid + g * 256;
        int kk  = lin >> 5;             // 0..31
        int mg  = (lin & 31) << 2;      // 0,4,...,124
        int k   = k0 + kk;
        float4 v = make_float4(0.f, 0.f, 0.f, 0.f);
        if (k < p.K && (m0 + mg) < p.M)
          v = *(const float4*)(A + (size_t)k * p.lda + (m0 + mg));
        stA[g] = v;
      }
    }
  };

  auto storeA = [&]() {
    if (AMODE == AM_ROW) {
      #pragma unroll
      for (int g = 0; g < 4; ++g) {
        int lin = tid + g * 256;
        int m   = lin >> 3;
        int kg  = (lin & 7) << 2;
        unsigned h0,h1,h2,h3,l0,l1,l2,l3;
        split1(stA[g].x,h0,l0); split1(stA[g].y,h1,l1);
        split1(stA[g].z,h2,l2); split1(stA[g].w,h3,l3);
        *(uint2*)(As_hi + m * LDS_STRIDE + kg) = make_uint2(h0 | (h1<<16), h2 | (h3<<16));
        *(uint2*)(As_lo + m * LDS_STRIDE + kg) = make_uint2(l0 | (l1<<16), l2 | (l3<<16));
      }
    } else {
      #pragma unroll
      for (int g = 0; g < 4; ++g) {
        int lin = tid + g * 256;
        int kk  = lin >> 5;
        int mg  = (lin & 31) << 2;
        unsigned h, l;
        split1(stA[g].x,h,l); As_hi[(mg+0)*LDS_STRIDE+kk]=(unsigned short)h; As_lo[(mg+0)*LDS_STRIDE+kk]=(unsigned short)l;
        split1(stA[g].y,h,l); As_hi[(mg+1)*LDS_STRIDE+kk]=(unsigned short)h; As_lo[(mg+1)*LDS_STRIDE+kk]=(unsigned short)l;
        split1(stA[g].z,h,l); As_hi[(mg+2)*LDS_STRIDE+kk]=(unsigned short)h; As_lo[(mg+2)*LDS_STRIDE+kk]=(unsigned short)l;
        split1(stA[g].w,h,l); As_hi[(mg+3)*LDS_STRIDE+kk]=(unsigned short)h; As_lo[(mg+3)*LDS_STRIDE+kk]=(unsigned short)l;
      }
    }
  };

  auto fetchB = [&](int k0) {
    if (BMODE == BM_TRANS) {  // b(k,j) = B[j*ldb + k]  (k-contiguous source)
      #pragma unroll
      for (int g = 0; g < 4; ++g) {
        int lin = tid + g * 256;
        int j   = lin >> 3;
        int kg  = (lin & 7) << 2;
        int k   = k0 + kg;
        float4 v = make_float4(0.f, 0.f, 0.f, 0.f);
        if (k < p.K && (n0 + j) < p.N) {
          const float* src = Bp + (size_t)(n0 + j) * p.ldb + k;
          v = *(const float4*)src;
          if (g == 0 && k + 64 < p.K) __builtin_prefetch(src + 64, 0, 1);
        }
        stB[g] = v;
      }
    } else {  // BM_CONV: k=(c,t); b(k,j) = src[(j+t-1)*ldb + c], zero-padded
      #pragma unroll
      for (int g = 0; g < 16; ++g) {
        int lin = tid + g * 256;        // 0..4095
        int j   = lin >> 5;
        int kk  = lin & 31;
        int k   = k0 + kk;
        float v = 0.f;
        if (k < p.K) {
          int c  = k / 3;
          int t  = k - 3 * c;
          int js = n0 + j + t - 1;
          if (js >= 0 && js < p.Nsrc) v = Bp[(size_t)js * p.ldb + c];
        }
        stBc[g] = v;
      }
    }
  };

  auto storeB = [&]() {
    if (BMODE == BM_TRANS) {
      #pragma unroll
      for (int g = 0; g < 4; ++g) {
        int lin = tid + g * 256;
        int j   = lin >> 3;
        int kg  = (lin & 7) << 2;
        unsigned h0,h1,h2,h3,l0,l1,l2,l3;
        split1(stB[g].x,h0,l0); split1(stB[g].y,h1,l1);
        split1(stB[g].z,h2,l2); split1(stB[g].w,h3,l3);
        *(uint2*)(Bs_hi + j * LDS_STRIDE + kg) = make_uint2(h0 | (h1<<16), h2 | (h3<<16));
        *(uint2*)(Bs_lo + j * LDS_STRIDE + kg) = make_uint2(l0 | (l1<<16), l2 | (l3<<16));
      }
    } else {
      #pragma unroll
      for (int g = 0; g < 16; ++g) {
        int lin = tid + g * 256;
        int j   = lin >> 5;
        int kk  = lin & 31;
        unsigned h, l; split1(stBc[g], h, l);
        Bs_hi[j * LDS_STRIDE + kk] = (unsigned short)h;
        Bs_lo[j * LDS_STRIDE + kk] = (unsigned short)l;
      }
    }
  };

  fetchA(0);
  fetchB(0);

  for (int kt = 0; kt < nk; ++kt) {
    // previous compute finished behind the barrier at loop bottom: LDS is free
    storeA();
    storeB();
    __syncthreads();

    // issue next iteration's global loads; latency overlaps with WMMA below
    if (kt + 1 < nk) {
      fetchA((kt + 1) << 5);
      fetchB((kt + 1) << 5);
    }

    // ---------------- compute: 24 WMMAs per wave per k-step --------------
    v16bf bh[4], bl[4];
    #pragma unroll
    for (int nt = 0; nt < 4; ++nt) {
      int col = wn * 64 + nt * 16 + row16;
      bh[nt] = ldfrag(Bs_hi, col, ksel);
      bl[nt] = ldfrag(Bs_lo, col, ksel);
    }
    #pragma unroll
    for (int mt = 0; mt < 2; ++mt) {
      int r = wm * 32 + mt * 16 + row16;
      v16bf ah = ldfrag(As_hi, r, ksel);
      v16bf al = ldfrag(As_lo, r, ksel);
      #pragma unroll
      for (int nt = 0; nt < 4; ++nt) {
        acc[mt][nt] = wmma_bf16(ah, bh[nt], acc[mt][nt]);
        acc[mt][nt] = wmma_bf16(ah, bl[nt], acc[mt][nt]);
        acc[mt][nt] = wmma_bf16(al, bh[nt], acc[mt][nt]);
      }
    }
    __syncthreads();
  }

  // ---------------- epilogue: bias + guarded stores ----------------------
  const int half = lane >> 4;
  #pragma unroll
  for (int mt = 0; mt < 2; ++mt) {
    #pragma unroll
    for (int nt = 0; nt < 4; ++nt) {
      int n = n0 + wn * 64 + nt * 16 + row16;
      float bn = 0.f;
      if (BIASMODE == BIAS_N && n < p.N) bn = p.bias[n];
      #pragma unroll
      for (int r = 0; r < 8; ++r) {
        int m = m0 + wm * 32 + mt * 16 + half * 8 + r;
        if (m < p.M && n < p.N) {
          float v = acc[mt][nt][r];
          if (BIASMODE == BIAS_N) v += bn;
          if (BIASMODE == BIAS_M) v += p.bias[m];
          C[(size_t)m * p.ldc + n] = v;
        }
      }
    }
  }
}

// row-wise squared L2 norm (one block per row)
__global__ __launch_bounds__(256)
void rownorm_kernel(const float* __restrict__ X, float* __restrict__ out, int ncols) {
  size_t row = blockIdx.x;
  const float* x = X + row * (size_t)ncols;
  float s = 0.f;
  for (int c = threadIdx.x; c < ncols; c += 256) { float v = x[c]; s += v * v; }
  for (int off = 16; off; off >>= 1) s += __shfl_xor(s, off, 32);
  __shared__ float sh[8];
  if ((threadIdx.x & 31) == 0) sh[threadIdx.x >> 5] = s;
  __syncthreads();
  if (threadIdx.x == 0) {
    float t = 0.f;
    #pragma unroll
    for (int w = 0; w < 8; ++w) t += sh[w];
    out[row] = t;
  }
}

// fused: -cdist + beta-binomial prior + row log-softmax (one block per (b,i))
__global__ __launch_bounds__(256)
void softmax_prior_kernel(const float* __restrict__ G, const float* __restrict__ qn,
                          const float* __restrict__ kn, const int* __restrict__ qlens,
                          const int* __restrict__ klens, float* __restrict__ out,
                          int TQ, int TK) {
  const int b = blockIdx.y, i = blockIdx.x, tid = threadIdx.x;
  const size_t rowoff = ((size_t)b * TQ + i) * TK;
  const float ql = (float)qlens[b];
  const float kl = (float)klens[b];
  const float fi = (float)i;
  const float a  = fi + 1.0f;
  const float bb = fmaxf(ql - fi, 1.0f);
  const float n  = fmaxf(kl - 1.0f, 0.0f);
  const float rowc = lgammaf(n + 1.0f) - (lgammaf(a) + lgammaf(bb) - lgammaf(a + bb));
  const float qv = qn[(size_t)b * TQ + i];
  const bool  rowvalid = fi < ql;

  float logit[4];
  float lmax = -3.0e38f;
  #pragma unroll
  for (int s = 0; s < 4; ++s) {
    int j = tid + s * 256;
    float jf  = (float)j;
    float nmj = fmaxf(n - jf, 0.0f);
    float lp  = rowc - lgammaf(jf + 1.0f) - lgammaf(nmj + 1.0f)
              + lgammaf(jf + a) + lgammaf(nmj + bb) - lgammaf(jf + a + nmj + bb);
    float prior = (rowvalid && (jf < kl)) ? lp : -1.0e9f;
    float sq = qv + kn[(size_t)b * TK + j] - 2.0f * G[rowoff + j];
    float d  = -sqrtf(fmaxf(sq, 1e-12f));
    logit[s] = d + prior;
    lmax = fmaxf(lmax, logit[s]);
  }

  __shared__ float sh[8];
  float v = lmax;
  for (int off = 16; off; off >>= 1) v = fmaxf(v, __shfl_xor(v, off, 32));
  if ((tid & 31) == 0) sh[tid >> 5] = v;
  __syncthreads();
  float gmax = sh[0];
  #pragma unroll
  for (int w = 1; w < 8; ++w) gmax = fmaxf(gmax, sh[w]);
  __syncthreads();

  float ls = 0.f;
  #pragma unroll
  for (int s = 0; s < 4; ++s) ls += expf(logit[s] - gmax);
  for (int off = 16; off; off >>= 1) ls += __shfl_xor(ls, off, 32);
  if ((tid & 31) == 0) sh[tid >> 5] = ls;
  __syncthreads();
  float gsum = 0.f;
  #pragma unroll
  for (int w = 0; w < 8; ++w) gsum += sh[w];
  float lse = gmax + logf(gsum);

  #pragma unroll
  for (int s = 0; s < 4; ++s) {
    int j = tid + s * 256;
    out[rowoff + j] = logit[s] - lse;
  }
}

extern "C" void kernel_launch(void* const* d_in, const int* in_sizes, int n_in,
                              void* d_out, int out_size, void* d_ws, size_t ws_size,
                              hipStream_t stream) {
  const int  BN = 16;
  const int  NQ = 2044, NK = 1024, QDIM = 80, H = 1024;
  const int  TQ = 2 * H;        // 2048
  const int  NQ1 = NQ + 2;      // 2046

  const float* query = (const float*)d_in[0];
  const float* key   = (const float*)d_in[1];
  const int*   qlens = (const int*)  d_in[2];
  const int*   klens = (const int*)  d_in[3];
  const float* qc1_w = (const float*)d_in[4];
  const float* qc1_b = (const float*)d_in[5];
  const float* qc2_w = (const float*)d_in[6];
  const float* qc2_b = (const float*)d_in[7];
  const float* ql_w  = (const float*)d_in[8];
  const float* ql_b  = (const float*)d_in[9];
  const float* kc_w  = (const float*)d_in[10];
  const float* kc_b  = (const float*)d_in[11];
  const float* kl1_w = (const float*)d_in[12];
  const float* kl1_b = (const float*)d_in[13];
  const float* kl2_w = (const float*)d_in[14];
  const float* kl2_b = (const float*)d_in[15];

  float* ws = (float*)d_ws;
  // workspace layout (floats), with buffer reuse:
  const size_t n_A1t = (size_t)BN * NQ1 * TQ;        // 67,043,328  (later reused for G)
  const size_t n_A2t = (size_t)BN * TQ * TQ;         // 67,108,864  (later K2 @ +0, Kf @ +33.5M)
  const size_t n_Qf  = (size_t)BN * TQ * H;          // 33,554,432
  const size_t n_K1  = (size_t)BN * H * H;           // 16,777,216
  float* A1t = ws;
  float* A2t = A1t + n_A1t;
  float* Qf  = A2t + n_A2t;
  float* K1  = Qf  + n_Qf;
  float* qnb = K1  + n_K1;                            // BN*TQ
  float* knb = qnb + (size_t)BN * TQ;                 // BN*H
  float* K2  = A2t;                                   // reuse (A2t dead after Q3)
  float* Kf  = A2t + (size_t)BN * H * (2 * H);        // reuse tail of A2t block
  float* G   = A1t;                                   // reuse (A1t dead after Q2)

  dim3 blk(256);
  GemmParams p;

  // Q1: A1t[n][o] = sum_c query_pad[n,c] * qc1_w[o,c] + qc1_b[o]
  p = { query, qc1_w, qc1_b, A1t, NQ1, TQ, QDIM, QDIM, QDIM, TQ,
        -1, NQ, 0, (long long)NQ * QDIM, 0, (long long)NQ1 * TQ };
  gemm_kernel<AM_ROW, BM_TRANS, BIAS_N><<<dim3(TQ/128, (NQ1+127)/128, BN), blk, 0, stream>>>(p);

  // Q2: A2t[n][o] = sum_c A1t_pad[n,c] * qc2_w[o,c] + qc2_b[o]
  p = { A1t, qc2_w, qc2_b, A2t, TQ, TQ, TQ, TQ, TQ, TQ,
        -1, NQ1, 0, (long long)NQ1 * TQ, 0, (long long)TQ * TQ };
  gemm_kernel<AM_ROW, BM_TRANS, BIAS_N><<<dim3(TQ/128, TQ/128, BN), blk, 0, stream>>>(p);

  // Q3: Qf[c][h] = sum_n A2t[n][c] * ql_w[h,n] + ql_b[h]   (A transposed source)
  p = { A2t, ql_w, ql_b, Qf, TQ, H, TQ, TQ, TQ, H,
        0, TQ, 0, (long long)TQ * TQ, 0, (long long)TQ * H };
  gemm_kernel<AM_TRANS, BM_TRANS, BIAS_N><<<dim3(H/128, TQ/128, BN), blk, 0, stream>>>(p);

  // K1: K1[o][n] = conv_k3(key)[o,n] + kc_b[o]
  p = { kc_w, key, kc_b, K1, H, H, 768, 768, 256, H,
        0, H, NK, 0, (long long)NK * 256, (long long)H * H };
  gemm_kernel<AM_ROW, BM_CONV, BIAS_M><<<dim3(H/128, H/128, BN), blk, 0, stream>>>(p);

  // K2: K2[c][o2] = sum_n K1[c,n] * kl1_w[o2,n] + kl1_b[o2]
  p = { K1, kl1_w, kl1_b, K2, H, 2*H, H, H, H, 2*H,
        0, H, 0, (long long)H * H, 0, (long long)H * 2 * H };
  gemm_kernel<AM_ROW, BM_TRANS, BIAS_N><<<dim3((2*H)/128, H/128, BN), blk, 0, stream>>>(p);

  // K3: Kf[c][h] = sum_o K2[c,o] * kl2_w[h,o] + kl2_b[h]
  p = { K2, kl2_w, kl2_b, Kf, H, H, 2*H, 2*H, 2*H, H,
        0, H, 0, (long long)H * 2 * H, 0, (long long)H * H };
  gemm_kernel<AM_ROW, BM_TRANS, BIAS_N><<<dim3(H/128, H/128, BN), blk, 0, stream>>>(p);

  // norms
  rownorm_kernel<<<dim3((unsigned)((size_t)BN * TQ)), blk, 0, stream>>>(Qf, qnb, H);
  rownorm_kernel<<<dim3((unsigned)((size_t)BN * H)),  blk, 0, stream>>>(Kf, knb, H);

  // DIST: G[n][m] = sum_h Qf[n,h] * Kf[m,h]
  p = { Qf, Kf, nullptr, G, TQ, H, H, H, H, H,
        0, TQ, 0, (long long)TQ * H, (long long)H * H, (long long)TQ * H };
  gemm_kernel<AM_ROW, BM_TRANS, BIAS_NONE><<<dim3(H/128, TQ/128, BN), blk, 0, stream>>>(p);

  // fused -cdist + prior + log_softmax
  softmax_prior_kernel<<<dim3(TQ, BN), blk, 0, stream>>>(G, qnb, knb, qlens, klens,
                                                         (float*)d_out, TQ, H);
}
